// DeepseekV2MoE_88235808129209
// MI455X (gfx1250) — matrix operand order repaired and checked by reading, not compiled, not run on hardware
//
#include <hip/hip_runtime.h>
#include <hip/hip_bf16.h>

// ---------------- problem constants (from reference setup_inputs) ----------
#define T_TOK  1024
#define HDIM   2048
#define IDIM   1408
#define NEXP   16
#define ISH    2816
#define NGROUP 4
#define GSIZE  (NEXP / NGROUP)   // 4 experts per group
#define TOPKG  2
#define TOPK   4
#define RSCALE 2.5f

typedef __bf16 bf16t;
typedef __attribute__((ext_vector_type(16))) __bf16 v16bf;
typedef __attribute__((ext_vector_type(8)))  __bf16 v8bf;
typedef __attribute__((ext_vector_type(4)))  __bf16 v4bf;
typedef __attribute__((ext_vector_type(2)))  __bf16 v2bf;
typedef __attribute__((ext_vector_type(8)))  float  v8f;

#define LDA  40   // A tile row stride (elems): 32 + 8 pad, 80B = 16B-aligned rows
#define LDBK 40   // B tile (transposed [n][k]) k-stride: 32 + 8 pad

__device__ __forceinline__ float fast_silu(float x) {
  // x * v_rcp_f32(1+exp(-x)) : avoids the IEEE div macro-expansion
  return x * __builtin_amdgcn_rcpf(1.0f + __expf(-x));
}

union FragU { v16bf f; v8bf h[2]; };

// ---- WMMA fragment builders per CDNA5 ISA 7.12.2 (wave32) -----------------
// A (16x32 bf16, row-major in LDS): lanes 0-15 row m=lane: halves[0..7]=K0..7,
// [8..15]=K16..23 ; lanes 16-31 row m=lane-16: K8..15 / K24..31.
// Both runs are contiguous in LDS -> two 16B ds_load_b128.
__device__ __forceinline__ v16bf load_a_frag(const bf16t* lA, int lane, int mi) {
  FragU u;
  const bf16t* row = lA + (mi * 16 + (lane & 15)) * LDA + ((lane < 16) ? 0 : 8);
  u.h[0] = *(const v8bf*)(row);
  u.h[1] = *(const v8bf*)(row + 16);
  return u.f;
}
// B (32x16 bf16) stored TRANSPOSED in LDS as [n][k]: lanes 0-15 read K0..7 /
// K16..23, lanes 16-31 read K8..15 / K24..31 (matches A) -> two 16B ds_load_b128.
__device__ __forceinline__ v16bf load_b_frag(const bf16t* lBt, int lane, int ni) {
  FragU u;
  const bf16t* row = lBt + (ni * 16 + (lane & 15)) * LDBK + ((lane < 16) ? 0 : 8);
  u.h[0] = *(const v8bf*)(row);
  u.h[1] = *(const v8bf*)(row + 16);
  return u.f;
}

__device__ __forceinline__ v8f wmma_bf16(v16bf a, v16bf b, v8f c) {
  return __builtin_amdgcn_wmma_f32_16x16x32_bf16(false, a, false, b,
                                                 (short)0, c, false, false);
}

// Stage one 32k x 64n fp32 tile -> LDS bf16 [n][k] (transposed), 256 threads.
// thread t: k rows 2*(t>>4), 2*(t>>4)+1 ; n cols (t&15)*4 .. +3
// -> 2 global_load_b128 + packed ds_store_b32 pairs per thread.
__device__ __forceinline__ void stage_b(const float* __restrict__ gB, int ldg,
                                        bf16t* __restrict__ lBt, int t) {
  const int k0 = (t >> 4) * 2;
  const int n0 = (t & 15) * 4;
  const float4 r0 = *(const float4*)(gB + (size_t)k0 * ldg + n0);
  const float4 r1 = *(const float4*)(gB + (size_t)(k0 + 1) * ldg + n0);
  v2bf p;
  p[0] = (bf16t)r0.x; p[1] = (bf16t)r1.x; *(v2bf*)(lBt + (n0 + 0) * LDBK + k0) = p;
  p[0] = (bf16t)r0.y; p[1] = (bf16t)r1.y; *(v2bf*)(lBt + (n0 + 1) * LDBK + k0) = p;
  p[0] = (bf16t)r0.z; p[1] = (bf16t)r1.z; *(v2bf*)(lBt + (n0 + 2) * LDBK + k0) = p;
  p[0] = (bf16t)r0.w; p[1] = (bf16t)r1.w; *(v2bf*)(lBt + (n0 + 3) * LDBK + k0) = p;
}

// ---------------- kernel 1: fp32 -> bf16 convert of x ----------------------
__global__ __launch_bounds__(256) void cvt_bf16_kernel(const float* __restrict__ src,
                                                       bf16t* __restrict__ dst, int n4) {
  int i = blockIdx.x * 256 + threadIdx.x;
  if (i < n4) {
    float4 v = *(const float4*)(src + (size_t)i * 4);
    v4bf o;
    o[0] = (bf16t)v.x; o[1] = (bf16t)v.y; o[2] = (bf16t)v.z; o[3] = (bf16t)v.w;
    *(v4bf*)(dst + (size_t)i * 4) = o;
  }
}

// ---------------- kernel 2: router (one block per token) -------------------
__global__ __launch_bounds__(256) void router_kernel(const float* __restrict__ x,
                                                     const float* __restrict__ gate_w,
                                                     const float* __restrict__ bias,
                                                     int* __restrict__ cnt,
                                                     int* __restrict__ list,
                                                     float* __restrict__ row_w) {
  const int t   = blockIdx.x;
  const int tid = threadIdx.x;
  __shared__ float part[NEXP][16];
  __shared__ float logit[NEXP];

  const int e   = tid & (NEXP - 1);
  const int l16 = tid >> 4;
  float s = 0.f;
  const float* xr = x + (size_t)t * HDIM;
  const float* gr = gate_w + (size_t)e * HDIM;
  for (int h = l16; h < HDIM; h += 16) s += xr[h] * gr[h];
  part[e][l16] = s;
  __syncthreads();
  if (tid < NEXP) {
    float acc = 0.f;
#pragma unroll
    for (int j = 0; j < 16; ++j) acc += part[tid][j];
    logit[tid] = acc;
  }
  __syncthreads();
  if (tid == 0) {
    float score[NEXP], sb[NEXP];
#pragma unroll
    for (int i = 0; i < NEXP; ++i) {
      score[i] = __builtin_amdgcn_rcpf(1.f + __expf(-logit[i]));
      sb[i]    = score[i] + bias[i];
    }
    float gsc[NGROUP];
#pragma unroll
    for (int g = 0; g < NGROUP; ++g) {
      float m1 = -INFINITY, m2 = -INFINITY;
#pragma unroll
      for (int j = 0; j < GSIZE; ++j) {
        float v = sb[g * GSIZE + j];
        if (v > m1)      { m2 = m1; m1 = v; }
        else if (v > m2) { m2 = v; }
      }
      gsc[g] = m1 + m2;
    }
    bool gsel[NGROUP] = {false, false, false, false};
    for (int k = 0; k < TOPKG; ++k) {
      int bi = 0; float bv = -INFINITY;
      for (int g = 0; g < NGROUP; ++g)
        if (!gsel[g] && gsc[g] > bv) { bv = gsc[g]; bi = g; }
      gsel[bi] = true;
    }
    float cand[NEXP];
#pragma unroll
    for (int i = 0; i < NEXP; ++i) cand[i] = gsel[i / GSIZE] ? sb[i] : -INFINITY;
    int ids[TOPK]; float wv[TOPK]; float wsum = 0.f;
    for (int k = 0; k < TOPK; ++k) {
      int bi = 0; float bv = -INFINITY;
      for (int i = 0; i < NEXP; ++i)
        if (cand[i] > bv) { bv = cand[i]; bi = i; }
      cand[bi] = -INFINITY;
      ids[k] = bi;
      wv[k]  = score[bi];
      wsum  += wv[k];
    }
    const float inv = RSCALE / wsum;   // renorm + routed scaling folded in
    for (int k = 0; k < TOPK; ++k) {
      int ei    = ids[k];
      int slot  = atomicAdd(&cnt[ei], 1);
      int rowid = (t << 2) | k;        // deterministic row identity
      list[ei * T_TOK + slot] = rowid;
      row_w[rowid] = wv[k] * inv;
    }
  }
}

// ---------------- kernel 3: routed gate_up grouped GEMM --------------------
__global__ __launch_bounds__(256) void expert_gateup_kernel(
    const bf16t* __restrict__ xb, const float* __restrict__ w_gate_up,
    const int* __restrict__ cnt, const int* __restrict__ list,
    const float* __restrict__ row_w, bf16t* __restrict__ act) {
  const int e  = blockIdx.y >> 5;
  const int mt = blockIdx.y & 31;
  const int M  = cnt[e];
  const int M0 = mt * 32;
  if (M0 >= M) return;                 // uniform early-exit
  const int ci = blockIdx.x * 64;      // base in [0, IDIM)

  __shared__ __align__(16) bf16t lA[2 * 32 * LDA];
  __shared__ __align__(16) bf16t lBg[2 * 64 * LDBK];
  __shared__ __align__(16) bf16t lBu[2 * 64 * LDBK];
  __shared__ int   s_rowid[32];
  __shared__ int   s_tok[32];
  __shared__ float s_w[32];

  const int tid  = threadIdx.x;
  const int lane = tid & 31;
  const int wv   = tid >> 5;
  const int mi   = wv >> 2;   // 0..1
  const int ni   = wv & 3;    // 0..3

  if (tid < 32) {
    int gm = M0 + tid;
    if (gm < M) {
      int rowid    = list[e * T_TOK + gm];
      s_rowid[tid] = rowid;
      s_tok[tid]   = rowid >> 2;
      s_w[tid]     = row_w[rowid];
    } else {
      s_rowid[tid] = -1; s_tok[tid] = 0; s_w[tid] = 0.f;
    }
  }
  __syncthreads();

  const float* wg_base = w_gate_up + (size_t)e * HDIM * (2 * IDIM) + ci;
  const int ar = tid >> 3, ac = (tid & 7) * 4;
  // loop-invariant gathered A source pointer (one LDS read, hoisted)
  const bf16t* a_src = xb + (size_t)s_tok[ar] * HDIM + ac;

  auto stage = [&](int kk, int buf) {
    v4bf a4 = *(const v4bf*)(a_src + kk);
    *(v4bf*)(lA + buf * (32 * LDA) + ar * LDA + ac) = a4;
    const float* gB = wg_base + (size_t)kk * (2 * IDIM);
    stage_b(gB,        2 * IDIM, lBg + buf * (64 * LDBK), tid);
    stage_b(gB + IDIM, 2 * IDIM, lBu + buf * (64 * LDBK), tid);
  };

  v8f acc_g = {}; v8f acc_u = {};
  stage(0, 0);
  int buf = 0;
  for (int kk = 0; kk < HDIM; kk += 32) {
    __syncthreads();
    if (kk + 32 < HDIM) stage(kk + 32, buf ^ 1);   // overlap HBM with WMMA
    v16bf af = load_a_frag(lA  + buf * (32 * LDA),  lane, mi);
    v16bf bg = load_b_frag(lBg + buf * (64 * LDBK), lane, ni);
    v16bf bu = load_b_frag(lBu + buf * (64 * LDBK), lane, ni);
    acc_g = wmma_bf16(af, bg, acc_g);
    acc_u = wmma_bf16(af, bu, acc_u);
    buf ^= 1;
  }

  __align__(32) float gg[8], uu[8];
  *(v8f*)gg = acc_g; *(v8f*)uu = acc_u;
  const int nlocal = lane & 15;
  const int mbase  = mi * 16 + ((lane < 16) ? 0 : 8);
#pragma unroll
  for (int j = 0; j < 8; ++j) {
    int rl    = mbase + j;
    int rowid = s_rowid[rl];
    if (rowid >= 0) {
      float v = fast_silu(gg[j]) * uu[j] * s_w[rl];
      act[(size_t)rowid * IDIM + ci + ni * 16 + nlocal] = (bf16t)v;
    }
  }
}

// ---------------- kernel 4: routed down grouped GEMM -----------------------
__global__ __launch_bounds__(256) void expert_down_kernel(
    const bf16t* __restrict__ act, const float* __restrict__ w_down,
    const int* __restrict__ cnt, const int* __restrict__ list,
    float* __restrict__ down_buf) {
  const int e  = blockIdx.y >> 5;
  const int mt = blockIdx.y & 31;
  const int M  = cnt[e];
  const int M0 = mt * 32;
  if (M0 >= M) return;
  const int cn = blockIdx.x * 64;      // H column base

  __shared__ __align__(16) bf16t lA[2 * 32 * LDA];
  __shared__ __align__(16) bf16t lB[2 * 64 * LDBK];
  __shared__ int s_rowid[32];
  __shared__ int s_arow[32];

  const int tid  = threadIdx.x;
  const int lane = tid & 31;
  const int wv   = tid >> 5;
  const int mi   = wv >> 2;
  const int ni   = wv & 3;

  if (tid < 32) {
    int gm = M0 + tid;
    if (gm < M) {
      int rowid    = list[e * T_TOK + gm];
      s_rowid[tid] = rowid;
      s_arow[tid]  = rowid;
    } else {
      s_rowid[tid] = -1; s_arow[tid] = 0;
    }
  }
  __syncthreads();

  const float* wd_base = w_down + (size_t)e * IDIM * HDIM + cn;
  const int ar = tid >> 3, ac = (tid & 7) * 4;
  const bf16t* a_src = act + (size_t)s_arow[ar] * IDIM + ac;   // hoisted

  auto stage = [&](int kk, int buf) {
    v4bf a4 = *(const v4bf*)(a_src + kk);
    *(v4bf*)(lA + buf * (32 * LDA) + ar * LDA + ac) = a4;
    stage_b(wd_base + (size_t)kk * HDIM, HDIM, lB + buf * (64 * LDBK), tid);
  };

  v8f acc = {};
  stage(0, 0);
  int buf = 0;
  for (int kk = 0; kk < IDIM; kk += 32) {
    __syncthreads();
    if (kk + 32 < IDIM) stage(kk + 32, buf ^ 1);
    v16bf af = load_a_frag(lA + buf * (32 * LDA),  lane, mi);
    v16bf bf = load_b_frag(lB + buf * (64 * LDBK), lane, ni);
    acc = wmma_bf16(af, bf, acc);
    buf ^= 1;
  }

  __align__(32) float cc[8];
  *(v8f*)cc = acc;
  const int nlocal = lane & 15;
  const int mbase  = mi * 16 + ((lane < 16) ? 0 : 8);
#pragma unroll
  for (int j = 0; j < 8; ++j) {
    int rl    = mbase + j;
    int rowid = s_rowid[rl];
    if (rowid >= 0)
      down_buf[(size_t)rowid * HDIM + cn + ni * 16 + nlocal] = cc[j];
  }
}

// ---------------- kernel 5: shared expert gate_up GEMM ---------------------
__global__ __launch_bounds__(256) void shared_gateup_kernel(
    const bf16t* __restrict__ xb, const float* __restrict__ sh_gu,
    bf16t* __restrict__ s_act) {
  const int M0 = blockIdx.y * 32;
  const int ci = blockIdx.x * 64;      // base in [0, ISH)

  __shared__ __align__(16) bf16t lA[2 * 32 * LDA];
  __shared__ __align__(16) bf16t lBg[2 * 64 * LDBK];
  __shared__ __align__(16) bf16t lBu[2 * 64 * LDBK];

  const int tid  = threadIdx.x;
  const int lane = tid & 31;
  const int wv   = tid >> 5;
  const int mi   = wv >> 2;
  const int ni   = wv & 3;

  const float* gu_base = sh_gu + ci;
  const int ar = tid >> 3, ac = (tid & 7) * 4;
  const bf16t* a_src = xb + (size_t)(M0 + ar) * HDIM + ac;

  auto stage = [&](int kk, int buf) {
    v4bf a4 = *(const v4bf*)(a_src + kk);
    *(v4bf*)(lA + buf * (32 * LDA) + ar * LDA + ac) = a4;
    const float* gB = gu_base + (size_t)kk * (2 * ISH);
    stage_b(gB,       2 * ISH, lBg + buf * (64 * LDBK), tid);
    stage_b(gB + ISH, 2 * ISH, lBu + buf * (64 * LDBK), tid);
  };

  v8f acc_g = {}; v8f acc_u = {};
  stage(0, 0);
  int buf = 0;
  for (int kk = 0; kk < HDIM; kk += 32) {
    __syncthreads();
    if (kk + 32 < HDIM) stage(kk + 32, buf ^ 1);
    v16bf af = load_a_frag(lA  + buf * (32 * LDA),  lane, mi);
    v16bf bg = load_b_frag(lBg + buf * (64 * LDBK), lane, ni);
    v16bf bu = load_b_frag(lBu + buf * (64 * LDBK), lane, ni);
    acc_g = wmma_bf16(af, bg, acc_g);
    acc_u = wmma_bf16(af, bu, acc_u);
    buf ^= 1;
  }

  __align__(32) float gg[8], uu[8];
  *(v8f*)gg = acc_g; *(v8f*)uu = acc_u;
  const int nlocal = lane & 15;
  const int mbase  = mi * 16 + ((lane < 16) ? 0 : 8);
#pragma unroll
  for (int j = 0; j < 8; ++j) {
    int t = M0 + mbase + j;
    float v = fast_silu(gg[j]) * uu[j];
    s_act[(size_t)t * ISH + ci + ni * 16 + nlocal] = (bf16t)v;
  }
}

// ---------------- kernel 6: shared down GEMM + routed combine --------------
__global__ __launch_bounds__(256) void shared_down_combine_kernel(
    const bf16t* __restrict__ s_act, const float* __restrict__ sh_dn,
    const float* __restrict__ down_buf, float* __restrict__ out) {
  const int M0 = blockIdx.y * 32;
  const int cn = blockIdx.x * 64;

  __shared__ __align__(16) bf16t lA[2 * 32 * LDA];
  __shared__ __align__(16) bf16t lB[2 * 64 * LDBK];

  const int tid  = threadIdx.x;
  const int lane = tid & 31;
  const int wv   = tid >> 5;
  const int mi   = wv >> 2;
  const int ni   = wv & 3;

  const float* dn_base = sh_dn + cn;
  const int ar = tid >> 3, ac = (tid & 7) * 4;
  const bf16t* a_src = s_act + (size_t)(M0 + ar) * ISH + ac;

  auto stage = [&](int kk, int buf) {
    v4bf a4 = *(const v4bf*)(a_src + kk);
    *(v4bf*)(lA + buf * (32 * LDA) + ar * LDA + ac) = a4;
    stage_b(dn_base + (size_t)kk * HDIM, HDIM, lB + buf * (64 * LDBK), tid);
  };

  v8f acc = {};
  stage(0, 0);
  int buf = 0;
  for (int kk = 0; kk < ISH; kk += 32) {
    __syncthreads();
    if (kk + 32 < ISH) stage(kk + 32, buf ^ 1);
    v16bf af = load_a_frag(lA + buf * (32 * LDA),  lane, mi);
    v16bf bf = load_b_frag(lB + buf * (64 * LDBK), lane, ni);
    acc = wmma_bf16(af, bf, acc);
    buf ^= 1;
  }

  __align__(32) float cc[8];
  *(v8f*)cc = acc;
  const int nlocal = lane & 15;
  const int mbase  = mi * 16 + ((lane < 16) ? 0 : 8);
#pragma unroll
  for (int j = 0; j < 8; ++j) {
    int t   = M0 + mbase + j;
    int col = cn + ni * 16 + nlocal;
    const float* db = down_buf + ((size_t)t * 4) * HDIM + col;
    float v = cc[j] + db[0] + db[(size_t)HDIM] + db[2 * (size_t)HDIM] + db[3 * (size_t)HDIM];
    out[(size_t)t * HDIM + col] = v;
  }
}

// ---------------- host launcher --------------------------------------------
extern "C" void kernel_launch(void* const* d_in, const int* in_sizes, int n_in,
                              void* d_out, int out_size, void* d_ws, size_t ws_size,
                              hipStream_t stream) {
  const float* x      = (const float*)d_in[0];  // [T, H]
  const float* gate_w = (const float*)d_in[1];  // [E, H]
  const float* bias   = (const float*)d_in[2];  // [E]
  const float* w_gu   = (const float*)d_in[3];  // [E, H, 2I]
  const float* w_dn   = (const float*)d_in[4];  // [E, I, H]
  const float* sh_gu  = (const float*)d_in[5];  // [H, 2*ISH]
  const float* sh_dn  = (const float*)d_in[6];  // [ISH, H]
  float* out = (float*)d_out;
  (void)in_sizes; (void)n_in; (void)out_size; (void)ws_size;

  // workspace carve (~55 MB total)
  size_t off = 0;
  auto carve = [&](size_t bytes) -> char* {
    char* p = (char*)d_ws + off;
    off += (bytes + 255) & ~(size_t)255;
    return p;
  };
  bf16t* xb   = (bf16t*)carve((size_t)T_TOK * HDIM * sizeof(bf16t));       // 4 MB
  int*   cnt  = (int*)  carve(NEXP * sizeof(int));
  int*   list = (int*)  carve((size_t)NEXP * T_TOK * sizeof(int));         // 64 KB
  float* roww = (float*)carve((size_t)4 * T_TOK * sizeof(float));          // 16 KB
  bf16t* act  = (bf16t*)carve((size_t)4 * T_TOK * IDIM * sizeof(bf16t));   // 11.5 MB
  float* dbuf = (float*)carve((size_t)4 * T_TOK * HDIM * sizeof(float));   // 32 MB
  bf16t* sact = (bf16t*)carve((size_t)T_TOK * ISH * sizeof(bf16t));        // 5.6 MB

  hipMemsetAsync(cnt, 0, NEXP * sizeof(int), stream);

  cvt_bf16_kernel<<<(T_TOK * HDIM / 4 + 255) / 256, 256, 0, stream>>>(x, xb, T_TOK * HDIM / 4);
  router_kernel<<<T_TOK, 256, 0, stream>>>(x, gate_w, bias, cnt, list, roww);
  expert_gateup_kernel<<<dim3(IDIM / 64, NEXP * (T_TOK / 32)), 256, 0, stream>>>(
      xb, w_gu, cnt, list, roww, act);
  expert_down_kernel<<<dim3(HDIM / 64, NEXP * (T_TOK / 32)), 256, 0, stream>>>(
      act, w_dn, cnt, list, dbuf);
  shared_gateup_kernel<<<dim3(ISH / 64, T_TOK / 32), 256, 0, stream>>>(xb, sh_gu, sact);
  shared_down_combine_kernel<<<dim3(HDIM / 64, T_TOK / 32), 256, 0, stream>>>(
      sact, sh_dn, dbuf, out);
}
